// Quantization_80814104642233
// MI455X (gfx1250) — matrix operand order repaired and checked by reading, not compiled, not run on hardware
//
#include <hip/hip_runtime.h>

// ---------------------------------------------------------------------------
// VQ-VAE quantization for MI455X (gfx1250, wave32, WMMA + async-to-LDS).
//
// x[32,64,64,64] f32, emb[1024,64] f32 -> o (=emb gather), loss, indices.
// argmin_k ||z-e_k||^2 == argmax_k (z.e_k - 0.5||e_k||^2).
// Distance GEMM: 3xBF16-split WMMA (hi*hi + hi*lo + lo*hi), f32 accumulate.
// Codebook converted to bf16 hi/lo ONCE (prep kernel -> d_ws); main kernel
// streams chunks into LDS with GLOBAL_LOAD_ASYNC_TO_LDS_B128 (ASYNCcnt).
// ---------------------------------------------------------------------------

#define BLOCK   128        // 4 waves
#define ROWS    64         // z-rows per block (64 | 4096 -> one b per block)
#define CCH     64         // channels
#define KCODES  1024
#define CHUNK   128        // codes staged in LDS at a time
#define NCHUNKS (KCODES / CHUNK)
#define TILES_PER_CHUNK (CHUNK / 16)
#define NBLOCKS 2048       // 131072 / ROWS
#define NUMEL   8388608    // 32*64*64*64
#define LOSS_OFF 8388608
#define IDX_OFF  8388609

typedef __attribute__((ext_vector_type(16))) __bf16        v16bf;
typedef __attribute__((ext_vector_type(8)))  float         v8f;
typedef __attribute__((ext_vector_type(8)))  unsigned int  v8u;
// Exact pointee type of the async-copy builtin's pointer params (per hipcc):
typedef __attribute__((__vector_size__(4 * sizeof(int)))) int b128vec;

__device__ __forceinline__ unsigned short f2bf(float f) {   // round-to-nearest-even
  unsigned u = __float_as_uint(f);
  u += 0x7fffu + ((u >> 16) & 1u);
  return (unsigned short)(u >> 16);
}
__device__ __forceinline__ float bf2f(unsigned short h) {
  return __uint_as_float(((unsigned)h) << 16);
}
__device__ __forceinline__ v16bf pack16(uint4 a, uint4 b) {
  v8u u;
  u[0] = a.x; u[1] = a.y; u[2] = a.z; u[3] = a.w;
  u[4] = b.x; u[5] = b.y; u[6] = b.z; u[7] = b.w;
  return __builtin_bit_cast(v16bf, u);
}

// ---- CDNA5 async global->LDS copy (ASYNCcnt), with safe fallback ----------
__device__ __forceinline__ void async_copy16(const void* g, void* l) {
#if __has_builtin(__builtin_amdgcn_global_load_async_to_lds_b128)
  __builtin_amdgcn_global_load_async_to_lds_b128(
      (__attribute__((address_space(1))) b128vec*)const_cast<void*>(g),
      (__attribute__((address_space(3))) b128vec*)l, 0, 0);
#else
  *(uint4*)l = *(const uint4*)g;
#endif
}
__device__ __forceinline__ void wait_async_copies() {
#if __has_builtin(__builtin_amdgcn_s_wait_asynccnt)
  __builtin_amdgcn_s_wait_asynccnt(0);
#elif __has_builtin(__builtin_amdgcn_global_load_async_to_lds_b128)
  asm volatile("s_wait_asynccnt 0" ::: "memory");
#endif
}

// A operand (16x32 bf16, M=row, K=channel). Per ISA layout:
// lanes 0-15: VGPR0-3 = K cb..cb+7, VGPR4-7 = K cb+16..cb+23, cb=0
// lanes16-31: same with cb=8.
__device__ __forceinline__ v16bf load_a(const unsigned short* p, int row,
                                        int cbase, bool laneLow) {
  int cb = cbase + (laneLow ? 0 : 8);
  const unsigned short* base = p + row * CCH + cb;
  uint4 q0 = *(const uint4*)(base);
  uint4 q1 = *(const uint4*)(base + 16);
  return pack16(q0, q1);
}
// B operand (32x16 bf16, K=channel, N=code):
// lanes 0-15: K 0..7 and 16..23, lanes 16-31: K 8..15 and 24..31 (same as A).
__device__ __forceinline__ v16bf load_b(const unsigned short* p, int code,
                                        int cbase, bool laneLow) {
  int cb = cbase + (laneLow ? 0 : 8);
  const unsigned short* base = p + code * CCH + cb;
  uint4 q0 = *(const uint4*)(base);
  uint4 q1 = *(const uint4*)(base + 16);
  return pack16(q0, q1);
}

// ---- Prep: convert codebook to bf16 hi/lo + (-0.5*||e||^2), once ----------
__global__ __launch_bounds__(256) void vq_prep_kernel(
    const float* __restrict__ emb, unsigned short* __restrict__ whi,
    unsigned short* __restrict__ wlo, float* __restrict__ wnrm) {
  const int g = blockIdx.x * 256 + threadIdx.x;   // one code per thread
  if (g >= KCODES) return;
  float s = 0.f;
  const float4* e4 = (const float4*)(emb + g * CCH);
#pragma unroll 4
  for (int q = 0; q < CCH / 4; ++q) {
    float4 v = e4[q];
    float vv[4] = {v.x, v.y, v.z, v.w};
#pragma unroll
    for (int j = 0; j < 4; ++j) {
      int c = q * 4 + j;
      unsigned short h = f2bf(vv[j]);
      whi[g * CCH + c] = h;
      wlo[g * CCH + c] = f2bf(vv[j] - bf2f(h));
      s += vv[j] * vv[j];
    }
  }
  wnrm[g] = -0.5f * s;
}

__global__ __launch_bounds__(BLOCK) void vq_main_kernel(
    const float* __restrict__ x, const float* __restrict__ emb,
    const unsigned short* __restrict__ whi,
    const unsigned short* __restrict__ wlo,
    const float* __restrict__ wnrm,
    float* __restrict__ out, float* __restrict__ partials) {
  __shared__ __align__(16) unsigned short xs_hi[ROWS * CCH];   // 8 KB
  __shared__ __align__(16) unsigned short xs_lo[ROWS * CCH];   // 8 KB
  __shared__ __align__(16) unsigned short es_hi[CHUNK * CCH];  // 16 KB
  __shared__ __align__(16) unsigned short es_lo[CHUNK * CCH];  // 16 KB
  __shared__ __align__(16) float es_nrm[CHUNK];                // pre-scaled
  __shared__ int   idx_s[ROWS];
  __shared__ float redbuf[BLOCK];

  const int  t       = threadIdx.x;
  const int  lane    = t & 31;
  const int  wave    = t >> 5;
  const bool laneLow = lane < 16;

  const int n0  = blockIdx.x * ROWS;   // first row (b*4096 + hw)
  const int b   = n0 >> 12;
  const int hw0 = n0 & 4095;

  // ---- Phase 1: gather strided x tile [64 x 64], split into bf16 hi/lo
  {
    const int  row   = t & 63;
    const int  chalf = t >> 6;
    const long base  = ((long)b * CCH) << 12;
    const int  hw    = hw0 + row;
#pragma unroll 4
    for (int cc = 0; cc < 32; ++cc) {
      int   c = chalf * 32 + cc;
      float v = x[base + ((long)c << 12) + hw];   // 256B-coalesced across rows
      unsigned short h = f2bf(v);
      xs_hi[row * CCH + c] = h;
      xs_lo[row * CCH + c] = f2bf(v - bf2f(h));
    }
  }
  __syncthreads();

  // ---- Preload this wave's A operands (rows wave*16..+15), K=0..63
  const int arow = wave * 16 + (lane & 15);
  v16bf aH0 = load_a(xs_hi, arow, 0,  laneLow);
  v16bf aH1 = load_a(xs_hi, arow, 32, laneLow);
  v16bf aL0 = load_a(xs_lo, arow, 0,  laneLow);
  v16bf aL1 = load_a(xs_lo, arow, 32, laneLow);

  float best[8];
  int   bidx[8];
#pragma unroll
  for (int r = 0; r < 8; ++r) { best[r] = -1e30f; bidx[r] = 0; }

  for (int ch = 0; ch < NCHUNKS; ++ch) {
    __syncthreads();  // previous chunk's LDS readers are done
    // ---- Stage pre-converted bf16 chunk: pure global->LDS async copies
    {
      const uint4* ghi = (const uint4*)(whi + ch * CHUNK * CCH);
      const uint4* glo = (const uint4*)(wlo + ch * CHUNK * CCH);
      uint4* lhi = (uint4*)es_hi;
      uint4* llo = (uint4*)es_lo;
#pragma unroll
      for (int i = 0; i < 8; ++i) {          // 128 thr * 8 * 16B = 16 KB each
        int q = t + i * BLOCK;
        async_copy16(ghi + q, lhi + q);
        async_copy16(glo + q, llo + q);
      }
      if (t < CHUNK / 4)                      // 512 B of norms
        async_copy16((const uint4*)(wnrm + ch * CHUNK) + t,
                     (uint4*)es_nrm + t);
      wait_async_copies();                    // own wave's ASYNCcnt -> 0
    }
    __syncthreads();                          // all waves' copies visible

    // ---- 8 code tiles of 16: score = z.e - 0.5||e||^2 via 6 WMMAs each
#pragma unroll 2
    for (int tile = 0; tile < TILES_PER_CHUNK; ++tile) {
      const int   codeL = tile * 16 + (lane & 15);
      const int   codeG = ch * CHUNK + codeL;
      const float cinit = es_nrm[codeL];      // already -0.5*||e||^2
      v8f acc;
#pragma unroll
      for (int r = 0; r < 8; ++r) acc[r] = cinit;

      v16bf bH0 = load_b(es_hi, codeL, 0,  laneLow);
      v16bf bH1 = load_b(es_hi, codeL, 32, laneLow);
      v16bf bL0 = load_b(es_lo, codeL, 0,  laneLow);
      v16bf bL1 = load_b(es_lo, codeL, 32, laneLow);

      acc = __builtin_amdgcn_wmma_f32_16x16x32_bf16(false, aH0, false, bH0, (short)0, acc, false, false);
      acc = __builtin_amdgcn_wmma_f32_16x16x32_bf16(false, aH1, false, bH1, (short)0, acc, false, false);
      acc = __builtin_amdgcn_wmma_f32_16x16x32_bf16(false, aH0, false, bL0, (short)0, acc, false, false);
      acc = __builtin_amdgcn_wmma_f32_16x16x32_bf16(false, aH1, false, bL1, (short)0, acc, false, false);
      acc = __builtin_amdgcn_wmma_f32_16x16x32_bf16(false, aL0, false, bH0, (short)0, acc, false, false);
      acc = __builtin_amdgcn_wmma_f32_16x16x32_bf16(false, aL1, false, bH1, (short)0, acc, false, false);

      // ascending code order + strict '>' keeps the first winner on ties
#pragma unroll
      for (int r = 0; r < 8; ++r) {
        if (acc[r] > best[r]) { best[r] = acc[r]; bidx[r] = codeG; }
      }
    }
  }

  // ---- Cross-lane argmax: rows 0-7 live in lanes 0-15, rows 8-15 in 16-31.
#pragma unroll
  for (int r = 0; r < 8; ++r) {
    float v = best[r];
    int   i = bidx[r];
#pragma unroll
    for (int off = 8; off >= 1; off >>= 1) {
      float ov = __shfl_xor(v, off, 32);
      int   oi = __shfl_xor(i, off, 32);
      if (ov > v || (ov == v && oi < i)) { v = ov; i = oi; }
    }
    if ((lane & 15) == r)
      idx_s[wave * 16 + (laneLow ? r : 8 + r)] = i;
  }
  __syncthreads();

  // ---- Output: o = emb[idx] back to [B,C,H,W]; deterministic loss partial
  float lsum = 0.f;
  {
    const int  row   = t & 63;
    const int  chalf = t >> 6;
    const long base  = ((long)b * CCH) << 12;
    const int  hw    = hw0 + row;
    const int  code  = idx_s[row];
#pragma unroll 4
    for (int cc = 0; cc < 32; ++cc) {
      int   c = chalf * 32 + cc;
      float e = emb[code * CCH + c];          // fp32 gather, L2-hot
      long  a = base + ((long)c << 12) + hw;
      float d = x[a] - e;
      out[a] = e;                             // o = x + sg(e-x) == e
      lsum += d * d;
    }
  }
  if (t < ROWS) out[IDX_OFF + n0 + t] = (float)idx_s[t];

  redbuf[t] = lsum;
  __syncthreads();
  for (int s = BLOCK / 2; s > 0; s >>= 1) {
    if (t < s) redbuf[t] += redbuf[t + s];
    __syncthreads();
  }
  if (t == 0) partials[blockIdx.x] = redbuf[0];
}

// Deterministic fixed-order reduction of per-block partials -> scalar loss.
__global__ __launch_bounds__(256) void vq_finalize_kernel(
    const float* __restrict__ partials, float* __restrict__ out) {
  __shared__ float buf[256];
  float s = 0.f;
  for (int i = threadIdx.x; i < NBLOCKS; i += 256) s += partials[i];
  buf[threadIdx.x] = s;
  __syncthreads();
  for (int k = 128; k > 0; k >>= 1) {
    if (threadIdx.x < k) buf[threadIdx.x] += buf[threadIdx.x + k];
    __syncthreads();
  }
  if (threadIdx.x == 0) out[LOSS_OFF] = buf[0] * 1.25f / (float)NUMEL;  // (1+beta)*MSE
}

extern "C" void kernel_launch(void* const* d_in, const int* in_sizes, int n_in,
                              void* d_out, int out_size, void* d_ws, size_t ws_size,
                              hipStream_t stream) {
  (void)in_sizes; (void)n_in; (void)out_size; (void)ws_size;
  const float* x   = (const float*)d_in[0];
  const float* emb = (const float*)d_in[1];
  float* out = (float*)d_out;

  // Workspace layout (all rewritten every call; no cross-call state):
  //   [0,128K)   bf16-hi codebook   [128K,256K) bf16-lo codebook
  //   [256K,260K) -0.5*||e||^2      [260K,...)  per-block loss partials
  char* ws = (char*)d_ws;
  unsigned short* whi  = (unsigned short*)(ws);
  unsigned short* wlo  = (unsigned short*)(ws + 128 * 1024);
  float*          wnrm = (float*)(ws + 256 * 1024);
  float*          partials = (float*)(ws + 260 * 1024);

  vq_prep_kernel<<<(KCODES + 255) / 256, 256, 0, stream>>>(emb, whi, wlo, wnrm);
  vq_main_kernel<<<NBLOCKS, BLOCK, 0, stream>>>(x, emb, whi, wlo, wnrm, out, partials);
  vq_finalize_kernel<<<1, 256, 0, stream>>>(partials, out);
}
